// KipfGCN_49228915146887
// MI455X (gfx1250) — matrix operand order repaired and checked by reading, not compiled, not run on hardware
//
#include <hip/hip_runtime.h>

typedef __attribute__((ext_vector_type(16))) __bf16          v16bf;
typedef __attribute__((ext_vector_type(8)))  unsigned short  v8u16;
typedef __attribute__((ext_vector_type(8)))  float           v8f;

#define N_NODES 100000
#define N_EDGES 3200000
#define F_IN    512
#define F_HID   256
#define N_CLASS 41

#define LDS_STRIDE 40   // padded row stride (elems): 80B = 20 dwords -> conflict-free, 16B aligned

struct FragPair { v8u16 lo, hi; };

__device__ __forceinline__ unsigned short f2bf(float f) {
    unsigned int u = __float_as_uint(f);
    unsigned int r = u + 0x7FFFu + ((u >> 16) & 1u);   // round-to-nearest-even
    return (unsigned short)(r >> 16);
}
__device__ __forceinline__ unsigned int pack2bf(float a, float b) {
#if __has_builtin(__builtin_amdgcn_cvt_pk_bf16_f32)
    auto p = __builtin_amdgcn_cvt_pk_bf16_f32(a, b);   // v_cvt_pk_bf16_f32
    return __builtin_bit_cast(unsigned int, p);
#else
    return (unsigned int)f2bf(a) | ((unsigned int)f2bf(b) << 16);
#endif
}

// ---------------------------------------------------------------------------
// bf16 WMMA GEMM:  C[M,Nc] = A[M,K] * B[K,Nc]   (A,B f32 in memory, bf16 math)
// block = 256 threads (8 waves), tile = 128 rows x (COLT*16) cols, K step 32.
// Double-buffered LDS + register prefetch. loadRegs issues pure global loads
// (no consumer), so the loadcnt wait sinks past the WMMA chain into storeRegs;
// out-of-range zeroing happens there. Guards are clamp+select, branch-free.
// A tile LDS: row-major  [128][LDS_STRIDE] -> frag = 2x ds_load_b128
// B tile LDS: transposed [BN ][LDS_STRIDE] -> frag = 2x ds_load_b128
// ---------------------------------------------------------------------------
template <int COLT>
__global__ __launch_bounds__(256) void gcn_gemm_wmma(
    const float* __restrict__ A, const float* __restrict__ B,
    float* __restrict__ C, int M, int K, int Nc)
{
    constexpr int BN  = COLT * 16;
    constexpr int NBP = BN / 16;                 // B pairs per thread
    __shared__ unsigned short lAbuf[2][128 * LDS_STRIDE];
    __shared__ unsigned short lBbuf[2][BN * LDS_STRIDE];

    const int tid  = threadIdx.x;
    const int lane = tid & 31;
    const int wave = tid >> 5;
    const int half = lane >> 4;
    const int m    = lane & 15;
    const int row0 = blockIdx.x * 128;
    const int col0 = blockIdx.y * BN;

    // per-thread staging coordinates
    const int a_kp  = tid & 15;                  // k-pair within row
    const int a_r0  = tid >> 4;                  // first of 8 rows (stride 16)
    // per-thread fragment LDS offsets (elems)
    const int aoff  = (wave * 16 + m) * LDS_STRIDE + 8 * half;

    float2 pA[8];
    float2 pB[NBP];

    // ---- register prefetch: pure loads, clamped addresses, NO data use ----
    auto loadRegs = [&](int k0) {
#pragma unroll
        for (int i = 0; i < 8; ++i) {
            int row  = row0 + a_r0 + i * 16;
            int rowc = row < M - 1 ? row : M - 1;        // clamp (in-bounds)
            pA[i] = *(const float2*)&A[(size_t)rowc * K + (k0 + 2 * a_kp)];
        }
#pragma unroll
        for (int i = 0; i < NBP; ++i) {
            int p  = tid + i * 256;
            int c  = p % BN;
            int kp = p / BN;
            int col  = col0 + c;
            int colc = col < Nc - 1 ? col : Nc - 1;
            size_t base = (size_t)(k0 + 2 * kp) * Nc + colc;
            pB[i] = make_float2(B[base], B[base + Nc]);
        }
    };
    // ---- zero-select + convert + store into LDS buffer `buf` ----
    auto storeRegs = [&](int buf) {
#pragma unroll
        for (int i = 0; i < 8; ++i) {
            int r   = a_r0 + i * 16;
            int row = row0 + r;
            float2 v = pA[i];
            if (row >= M) { v.x = 0.0f; v.y = 0.0f; }    // select, no branch
            *(unsigned int*)&lAbuf[buf][r * LDS_STRIDE + 2 * a_kp] =
                pack2bf(v.x, v.y);
        }
#pragma unroll
        for (int i = 0; i < NBP; ++i) {
            int p  = tid + i * 256;
            int c  = p % BN;
            int kp = p / BN;
            float2 v = pB[i];
            if (col0 + c >= Nc) { v.x = 0.0f; v.y = 0.0f; }
            *(unsigned int*)&lBbuf[buf][c * LDS_STRIDE + 2 * kp] =
                pack2bf(v.x, v.y);
        }
    };

    v8f acc[COLT] = {};
    const int nch = K >> 5;

    loadRegs(0);
    storeRegs(0);
    __syncthreads();

    for (int c = 0; c < nch; ++c) {
        const int cur = c & 1;
        if (c + 1 < nch)
            loadRegs((c + 1) << 5);              // global loads in flight

        // ---- load all fragments, then run the WMMA chain ----
        FragPair fa, fb[COLT];
        const unsigned short* pa = &lAbuf[cur][aoff];
        fa.lo = *(const v8u16*)pa;               // ds_load_b128
        fa.hi = *(const v8u16*)(pa + 16);
#pragma unroll
        for (int t = 0; t < COLT; ++t) {
            const unsigned short* pb =
                &lBbuf[cur][(t * 16 + m) * LDS_STRIDE + 8 * half];
            fb[t].lo = *(const v8u16*)pb;
            fb[t].hi = *(const v8u16*)(pb + 16);
        }
        v16bf afrag = __builtin_bit_cast(v16bf, fa);
#pragma unroll
        for (int t = 0; t < COLT; ++t) {
            v16bf bfrag = __builtin_bit_cast(v16bf, fb[t]);
            acc[t] = __builtin_amdgcn_wmma_f32_16x16x32_bf16(
                false, afrag, false, bfrag, (short)0, acc[t], false, false);
        }

        if (c + 1 < nch) {
            __syncthreads();                     // all waves done reading nxt buf
            storeRegs(1 - cur);                  // loadcnt wait lands here
            __syncthreads();                     // nxt buf visible to all waves
        }
    }

    // ---- store D: VGPR r -> row (r + 8*half), col = lane&15 ----
    const bool fullMN = (row0 + 128 <= M) && (col0 + BN <= Nc);
    if (fullMN) {
#pragma unroll
        for (int t = 0; t < COLT; ++t)
#pragma unroll
            for (int r = 0; r < 8; ++r) {
                int row = row0 + wave * 16 + r + half * 8;
                int col = col0 + t * 16 + m;
                C[(size_t)row * Nc + col] = acc[t][r];
            }
    } else {
#pragma unroll
        for (int t = 0; t < COLT; ++t)
#pragma unroll
            for (int r = 0; r < 8; ++r) {
                int row = row0 + wave * 16 + r + half * 8;
                int col = col0 + t * 16 + m;
                if (row < M && col < Nc)
                    C[(size_t)row * Nc + col] = acc[t][r];
            }
    }
}

// ---------------------------------------------------------------------------
// Degree / normalization
// ---------------------------------------------------------------------------
__global__ void k_fill1(float* d, int n) {
    int i = blockIdx.x * blockDim.x + threadIdx.x;
    if (i < n) d[i] = 1.0f;                 // self loop contributes 1
}
__global__ void k_degscat(float* deg, const int* __restrict__ dst, int e) {
    int i = blockIdx.x * blockDim.x + threadIdx.x;
    if (i < e) atomicAdd(&deg[dst[i]], 1.0f);
}
__global__ void k_rsqrt(float* d, int n) {
    int i = blockIdx.x * blockDim.x + threadIdx.x;
    if (i < n) d[i] = rsqrtf(d[i]);         // deg >= 1
}

// ---------------------------------------------------------------------------
// F=256 path, float4-vectorized (64 threads per node)
// ---------------------------------------------------------------------------
__global__ void k_selfinit4(float* __restrict__ out, const float* __restrict__ h,
                            const float* __restrict__ dinv) {
    int node = blockIdx.x;
    int f4 = threadIdx.x;                   // 0..63
    float d = dinv[node];
    float dd = d * d;
    size_t idx = (size_t)node * F_HID + f4 * 4;
    float4 v = *(const float4*)&h[idx];
    v.x *= dd; v.y *= dd; v.z *= dd; v.w *= dd;
    *(float4*)&out[idx] = v;
}
__global__ void k_edgescat4(float* __restrict__ out, const float* __restrict__ h,
                            const float* __restrict__ dinv,
                            const int* __restrict__ src, const int* __restrict__ dst) {
    int e = blockIdx.x;
    int s = src[e];
    int d = dst[e];
    float norm = dinv[s] * dinv[d];
    int f4 = threadIdx.x;                   // 0..63
    float4 v = *(const float4*)&h[(size_t)s * F_HID + f4 * 4];
    float* o = &out[(size_t)d * F_HID + f4 * 4];
    atomicAdd(o + 0, norm * v.x);
    atomicAdd(o + 1, norm * v.y);
    atomicAdd(o + 2, norm * v.z);
    atomicAdd(o + 3, norm * v.w);
}
__global__ void k_biasrelu4(float* __restrict__ a, const float* __restrict__ b) {
    int node = blockIdx.x;
    int f4 = threadIdx.x;                   // 0..63
    size_t idx = (size_t)node * F_HID + f4 * 4;
    float4 v = *(const float4*)&a[idx];
    float4 bb = *(const float4*)&b[f4 * 4];
    v.x = fmaxf(v.x + bb.x, 0.0f);
    v.y = fmaxf(v.y + bb.y, 0.0f);
    v.z = fmaxf(v.z + bb.z, 0.0f);
    v.w = fmaxf(v.w + bb.w, 0.0f);
    *(float4*)&a[idx] = v;
}

// ---------------------------------------------------------------------------
// F=41 path (layer 2 aggregation into d_out)
// ---------------------------------------------------------------------------
__global__ void k_selfinit(float* __restrict__ out, const float* __restrict__ h,
                           const float* __restrict__ dinv, int F) {
    int node = blockIdx.x;
    int f = threadIdx.x;
    if (f < F) {
        float d = dinv[node];
        out[(size_t)node * F + f] = d * d * h[(size_t)node * F + f];
    }
}
__global__ void k_edgescat(float* __restrict__ out, const float* __restrict__ h,
                           const float* __restrict__ dinv,
                           const int* __restrict__ src, const int* __restrict__ dst,
                           int F) {
    int e = blockIdx.x;
    int s = src[e];
    int d = dst[e];
    float norm = dinv[s] * dinv[d];
    int f = threadIdx.x;
    if (f < F)
        atomicAdd(&out[(size_t)d * F + f], norm * h[(size_t)s * F + f]);
}

// ---------------------------------------------------------------------------
// Fused bias + log_softmax over 41 classes (in place on d_out)
// ---------------------------------------------------------------------------
__global__ void k_logsoftmax(float* __restrict__ out, const float* __restrict__ b, int n) {
    int i = blockIdx.x * blockDim.x + threadIdx.x;
    if (i >= n) return;
    float v[N_CLASS];
    float mx = -3.4e38f;
#pragma unroll
    for (int f = 0; f < N_CLASS; ++f) {
        v[f] = out[(size_t)i * N_CLASS + f] + b[f];
        mx = v[f] > mx ? v[f] : mx;
    }
    float s = 0.0f;
#pragma unroll
    for (int f = 0; f < N_CLASS; ++f) s += __expf(v[f] - mx);
    float lse = mx + __logf(s);
#pragma unroll
    for (int f = 0; f < N_CLASS; ++f)
        out[(size_t)i * N_CLASS + f] = v[f] - lse;
}

// ---------------------------------------------------------------------------
extern "C" void kernel_launch(void* const* d_in, const int* in_sizes, int n_in,
                              void* d_out, int out_size, void* d_ws, size_t ws_size,
                              hipStream_t stream) {
    const float* x  = (const float*)d_in[0];
    const int*   ei = (const int*)  d_in[1];   // [2, E] row-major
    const float* W1 = (const float*)d_in[2];
    const float* b1 = (const float*)d_in[3];
    const float* W2 = (const float*)d_in[4];
    const float* b2 = (const float*)d_in[5];
    float* out = (float*)d_out;
    float* ws  = (float*)d_ws;

    const int* src = ei;
    const int* dst = ei + N_EDGES;

    // workspace layout (floats): dinv | h1 | a1 | h2   (~222 MB total)
    float* dnorm = ws;
    float* h1 = dnorm + N_NODES;
    float* a1 = h1 + (size_t)N_NODES * F_HID;
    float* h2 = a1 + (size_t)N_NODES * F_HID;

    // degree^-1/2 (with self loops)
    k_fill1  <<<(N_NODES + 255) / 256, 256, 0, stream>>>(dnorm, N_NODES);
    k_degscat<<<(N_EDGES + 255) / 256, 256, 0, stream>>>(dnorm, dst, N_EDGES);
    k_rsqrt  <<<(N_NODES + 255) / 256, 256, 0, stream>>>(dnorm, N_NODES);

    // layer 1: h1 = x @ W1 (bf16 WMMA), aggregate, +b1, ReLU
    gcn_gemm_wmma<4><<<dim3((N_NODES + 127) / 128, F_HID / 64), 256, 0, stream>>>(
        x, W1, h1, N_NODES, F_IN, F_HID);
    k_selfinit4<<<N_NODES, 64, 0, stream>>>(a1, h1, dnorm);
    k_edgescat4<<<N_EDGES, 64, 0, stream>>>(a1, h1, dnorm, src, dst);
    k_biasrelu4<<<N_NODES, 64, 0, stream>>>(a1, b1);

    // layer 2: h2 = a1 @ W2 (41 cols padded to 48), aggregate into d_out
    gcn_gemm_wmma<3><<<dim3((N_NODES + 127) / 128, 1), 256, 0, stream>>>(
        a1, W2, h2, N_NODES, F_HID, N_CLASS);
    k_selfinit<<<N_NODES, 64, 0, stream>>>(out, h2, dnorm, N_CLASS);
    k_edgescat<<<N_EDGES, 64, 0, stream>>>(out, h2, dnorm, src, dst, N_CLASS);

    // +b2 then log_softmax, in place on d_out
    k_logsoftmax<<<(N_NODES + 255) / 256, 256, 0, stream>>>(out, b2, N_NODES);
}
